// PaiNN_25391846654193
// MI455X (gfx1250) — matrix operand order repaired and checked by reading, not compiled, not run on hardware
//
#include <hip/hip_runtime.h>
#include <hip/hip_bf16.h>

// ---------------------------------------------------------------------------
// PaiNN on MI455X (gfx1250, wave32).
//  - Node GEMMs via v_wmma_f32_16x16x32_bf16; wave tile 16x32 (2 acc sharing
//    one A fragment); B tile staged into LDS in *fragment-major* swizzled
//    order so each B fragment is two ds_load_b128 (no scalar u16 gathers).
//  - Edge phase: one wave per edge, RBF broadcast via readlane (no barriers
//    in hot loop); filters recomputed on the fly from d_ij; scatter via
//    global_atomic_add_f32 into L2-resident dq/dmu tables.
//  - Reciprocals in SiLU / normalize use v_rcp_f32 (no IEEE div expansion).
// ---------------------------------------------------------------------------

#define F_DIM   128
#define N_RBF   20
#define CUTOFF  5.0f
#define N_LAYERS 3
#define EPS_N   1e-8f

typedef __attribute__((ext_vector_type(16))) __bf16          v16bf;
typedef __attribute__((ext_vector_type(8)))  float           v8f;
typedef __attribute__((ext_vector_type(8)))  unsigned short  us8;

static __device__ __forceinline__ unsigned short f32_to_bf16(float f) {
    unsigned int u = __float_as_uint(f);
    u += 0x7FFFu + ((u >> 16) & 1u);       // round-to-nearest-even
    return (unsigned short)(u >> 16);
}

static __device__ __forceinline__ float fast_rcp(float x) {
    return __builtin_amdgcn_rcpf(x);       // v_rcp_f32, <=1 ulp
}

static __device__ __forceinline__ float silu(float v) {
    return v * fast_rcp(1.0f + __expf(-v));
}

// ---------------------------------------------------------------------------
// utility kernels
// ---------------------------------------------------------------------------
__global__ void zero_u32(unsigned int* p, size_t n) {
    size_t i = (size_t)blockIdx.x * blockDim.x + threadIdx.x;
    size_t s = (size_t)gridDim.x * blockDim.x;
    for (; i < n; i += s) p[i] = 0u;
}

__global__ void cvt_bf16(const float* __restrict__ src, unsigned short* __restrict__ dst, size_t n) {
    size_t i = (size_t)blockIdx.x * blockDim.x + threadIdx.x;
    size_t s = (size_t)gridDim.x * blockDim.x;
    for (; i < n; i += s) dst[i] = f32_to_bf16(src[i]);
}

__global__ void add_inplace(float* __restrict__ p, const float* __restrict__ a, int n) {
    int t = blockIdx.x * blockDim.x + threadIdx.x;
    if (t < n) p[t] += a[t];
}

// ---------------------------------------------------------------------------
// edge geometry: d_ij, dir_ij, cosine cutoff
// ---------------------------------------------------------------------------
__global__ void geom_kernel(const float* __restrict__ r, float* __restrict__ dlen,
                            float* __restrict__ fcut, float* __restrict__ dir, int E) {
    int e = blockIdx.x * blockDim.x + threadIdx.x;
    if (e >= E) return;
    float x = r[3 * e + 0], y = r[3 * e + 1], z = r[3 * e + 2];
    float d = sqrtf(x * x + y * y + z * z);
    float inv = fast_rcp(d);
    dlen[e] = d;
    dir[3 * e + 0] = x * inv;
    dir[3 * e + 1] = y * inv;
    dir[3 * e + 2] = z * inv;
    float fc = 0.5f * (__cosf(d * (3.14159265358979f / CUTOFF)) + 1.0f);
    fcut[e] = (d < CUTOFF) ? fc : 0.0f;
}

__global__ void embed_kernel(const float* __restrict__ emb, const int* __restrict__ z,
                             float* __restrict__ q, int N) {
    int t = blockIdx.x * blockDim.x + threadIdx.x;
    if (t >= N * F_DIM) return;
    int n = t >> 7, f = t & 127;
    q[t] = emb[(size_t)z[n] * F_DIM + f];
}

// ---------------------------------------------------------------------------
// WMMA GEMM: C[M,Nc] = act(A[M,K]_bf16 @ B[K,Nc]_bf16 + bias)
// block = 128 threads = 4 waves; wave -> 16(M) x 32(N) tile (2 accumulators,
// shared A fragment). Block tile: 64(M) x 32(N). B staged into LDS swizzled
// fragment-major: sB[((kstep*2+c)*32 + lane)*16 + e] -> 2x ds_load_b128/frag.
// M multiple of 64, Nc multiple of 32, K in {128,256}.
// ---------------------------------------------------------------------------
template <int K, int ACT>
__global__ __launch_bounds__(128) void wmma_gemm_kernel(
    const unsigned short* __restrict__ A, const unsigned short* __restrict__ B,
    const float* __restrict__ bias, float* __restrict__ C, int Nc)
{
    __shared__ __align__(16) unsigned short sB[K * 32];
    const int tid   = threadIdx.x;
    const int lane  = tid & 31;
    const int wave  = tid >> 5;
    const int col16 = lane & 15;
    const int hi    = lane >> 4;
    const int tn    = blockIdx.x * 32;
    const int tm    = (blockIdx.y * 4 + wave) * 16;

    // cooperative stage of B[K,32] tile into LDS, swizzled to fragment order:
    // (k, col) -> sB[((kstep*2 + c)*32 + hi16*16 + n)*16 + e]
    for (int i = tid; i < K * 32; i += 128) {
        int k = i >> 5, col = i & 31;
        int c = col >> 4, n = col & 15;
        int kstep = k >> 5, h = (k >> 3) & 1, e = (k & 7) + (((k >> 4) & 1) << 3);
        sB[((((kstep << 1) + c) << 5) + (h << 4) + n) * 16 + e] = B[(size_t)k * Nc + tn + col];
    }
    __syncthreads();

    float b0 = bias ? bias[tn + col16]      : 0.0f;
    float b1 = bias ? bias[tn + 16 + col16] : 0.0f;
    v8f acc0, acc1;
#pragma unroll
    for (int r = 0; r < 8; ++r) { acc0[r] = b0; acc1[r] = b1; }

    // A fragment: lane holds row (tm + col16); K runs [k0+hi*8 .. +7] and
    // [k0+16+hi*8 .. +7] -> two contiguous 16B global loads per k-step.
    const unsigned short* Ap = A + (size_t)(tm + col16) * K + hi * 8;
#pragma unroll
    for (int k0 = 0; k0 < K; k0 += 32) {
        union { v16bf v; us8 h[2]; } af;
        af.h[0] = *(const us8*)(Ap + k0);
        af.h[1] = *(const us8*)(Ap + k0 + 16);

        const unsigned short* bp0 = &sB[(((k0 >> 5) << 1) + 0) * 512 + lane * 16];
        const unsigned short* bp1 = bp0 + 512;
        union { v16bf v; us8 h[2]; } bf0, bf1;
        bf0.h[0] = *(const us8*)(bp0);
        bf0.h[1] = *(const us8*)(bp0 + 8);
        bf1.h[0] = *(const us8*)(bp1);
        bf1.h[1] = *(const us8*)(bp1 + 8);

        acc0 = __builtin_amdgcn_wmma_f32_16x16x32_bf16(
            false, af.v, false, bf0.v, (short)0, acc0, false, false);
        acc1 = __builtin_amdgcn_wmma_f32_16x16x32_bf16(
            false, af.v, false, bf1.v, (short)0, acc1, false, false);
    }

#pragma unroll
    for (int r = 0; r < 8; ++r) {
        int row = tm + hi * 8 + r;
        float v0 = acc0[r], v1 = acc1[r];
        if (ACT) { v0 = silu(v0); v1 = silu(v1); }
        C[(size_t)row * Nc + tn + col16]      = v0;
        C[(size_t)row * Nc + tn + 16 + col16] = v1;
    }
}

// ---------------------------------------------------------------------------
// edge message kernel: ONE WAVE PER EDGE; 4 features per lane; RBF values
// computed by lanes 0..19 and broadcast with __shfl (v_readlane). Filter
// weight slice for this layer cached in LDS (single entry barrier).
// ---------------------------------------------------------------------------
__global__ __launch_bounds__(128) void edge_kernel(
    const float* __restrict__ dlen, const float* __restrict__ fcut,
    const float* __restrict__ dir,
    const int* __restrict__ idx_i, const int* __restrict__ idx_j,
    const float* __restrict__ x, const float* __restrict__ mu,
    const float* __restrict__ filt_W, const float* __restrict__ filt_b,
    float* __restrict__ dq, float* __restrict__ dmu, int E, int layer)
{
    __shared__ float sW[N_RBF * 3 * F_DIM];   // 30720 B
    __shared__ float sb[3 * F_DIM];
    const int tid  = threadIdx.x;
    const int lane = tid & 31;
    const int wid  = (blockIdx.x << 2) + (tid >> 5);   // global wave id
    const int nwav = gridDim.x << 2;

    const int cols = 3 * F_DIM;               // 384
    const int WCOL = N_LAYERS * 3 * F_DIM;    // 1152
    for (int i = tid; i < N_RBF * cols; i += 128) {
        int r = i / cols, c = i - r * cols;
        sW[i] = filt_W[(size_t)r * WCOL + layer * cols + c];
    }
    for (int i = tid; i < cols; i += 128) sb[i] = filt_b[layer * cols + i];
    __syncthreads();

    const float del   = CUTOFF / (float)(N_RBF - 1);
    const float coeff = -0.5f / (del * del);

    for (int e = wid; e < E; e += nwav) {
        // per-lane RBF value (lanes 0..19), broadcast below via readlane
        float dv = dlen[e];
        float dd = dv - (float)lane * del;
        float myphi = (lane < N_RBF) ? __expf(coeff * dd * dd) : 0.0f;

        const float fc   = fcut[e];
        const float dirx = dir[3 * e + 0];
        const float diry = dir[3 * e + 1];
        const float dirz = dir[3 * e + 2];
        const int j = idx_j[e];
        const int i = idx_i[e];

        // filter matvec: [20] x [20,384], each lane owns features lane+32c
        float w0[4], w1[4], w2[4];
#pragma unroll
        for (int c = 0; c < 4; ++c) {
            int t = lane + (c << 5);
            w0[c] = sb[t]; w1[c] = sb[F_DIM + t]; w2[c] = sb[2 * F_DIM + t];
        }
#pragma unroll
        for (int r = 0; r < N_RBF; ++r) {
            float p = __shfl(myphi, r, 32);
            const float* wr = sW + r * cols;
#pragma unroll
            for (int c = 0; c < 4; ++c) {
                int t = lane + (c << 5);
                w0[c] += p * wr[t];
                w1[c] += p * wr[F_DIM + t];
                w2[c] += p * wr[2 * F_DIM + t];
            }
        }

        const float* xj = x + (size_t)j * cols;
        const float* mj = mu + (size_t)j * cols;
        float* dqi  = dq  + (size_t)i * F_DIM;
        float* dmui = dmu + (size_t)i * cols;
#pragma unroll
        for (int c = 0; c < 4; ++c) {
            int t = lane + (c << 5);
            float g0 = w0[c] * fc, g1 = w1[c] * fc, g2 = w2[c] * fc;
            float dqv   = g0 * xj[t];
            float dmuR  = g1 * xj[F_DIM + t];
            float dmumu = g2 * xj[2 * F_DIM + t];
            atomicAdd(&dqi[t], dqv);
            atomicAdd(&dmui[t],             dmuR * dirx + dmumu * mj[t]);
            atomicAdd(&dmui[F_DIM + t],     dmuR * diry + dmumu * mj[F_DIM + t]);
            atomicAdd(&dmui[2 * F_DIM + t], dmuR * dirz + dmumu * mj[2 * F_DIM + t]);
        }
    }
}

// ---------------------------------------------------------------------------
// mixing context: mu_Vn = sqrt(sum_d mu_V^2 + eps); ctx_bf16 = [q | mu_Vn]
// ---------------------------------------------------------------------------
__global__ void ctx_kernel(const float* __restrict__ q, const float* __restrict__ mumix,
                           unsigned short* __restrict__ ctx, int N) {
    int t = blockIdx.x * blockDim.x + threadIdx.x;
    if (t >= N * F_DIM) return;
    int n = t >> 7, f = t & 127;
    const float* mm = mumix + (size_t)n * 3 * 256;
    float s = EPS_N;
#pragma unroll
    for (int d = 0; d < 3; ++d) { float v = mm[d * 256 + f]; s += v * v; }
    float vn = sqrtf(s);
    ctx[(size_t)n * 256 + f]         = f32_to_bf16(q[t]);
    ctx[(size_t)n * 256 + F_DIM + f] = f32_to_bf16(vn);
}

// final per-atom update with gated vector channel
__global__ void update_kernel(float* __restrict__ q, float* __restrict__ mu,
                              const float* __restrict__ mumix,
                              const float* __restrict__ x2, int N) {
    int t = blockIdx.x * blockDim.x + threadIdx.x;
    if (t >= N * F_DIM) return;
    int n = t >> 7, f = t & 127;
    const float* mm = mumix + (size_t)n * 3 * 256;
    float s = 0.0f, muW[3];
#pragma unroll
    for (int d = 0; d < 3; ++d) {
        float v = mm[d * 256 + f];
        float w = mm[d * 256 + F_DIM + f];
        muW[d] = w;
        s += v * w;
    }
    const float* xr = x2 + (size_t)n * 384;
    float dq_i = xr[f], dmu_i = xr[F_DIM + f], dqmu_i = xr[2 * F_DIM + f];
    q[t] += dq_i + dqmu_i * s;
#pragma unroll
    for (int d = 0; d < 3; ++d)
        mu[((size_t)n * 3 + d) * F_DIM + f] += dmu_i * muW[d];
}

__global__ void copy_out_kernel(const float* __restrict__ q, const float* __restrict__ mu,
                                float* __restrict__ out, int N) {
    int t = blockIdx.x * blockDim.x + threadIdx.x;
    int nq = N * F_DIM;
    int tot = N * 512;
    if (t < nq) out[t] = q[t];
    else if (t < tot) out[t] = mu[t - nq];
}

// ---------------------------------------------------------------------------
// host side
// ---------------------------------------------------------------------------
static inline void launch_gemm(const unsigned short* A, const unsigned short* B,
                               const float* bias, float* C, int M, int K, int Nc,
                               int act, hipStream_t s) {
    dim3 g(Nc / 32, M / 64), b(128);
    if (K == 128) {
        if (act) hipLaunchKernelGGL((wmma_gemm_kernel<128, 1>), g, b, 0, s, A, B, bias, C, Nc);
        else     hipLaunchKernelGGL((wmma_gemm_kernel<128, 0>), g, b, 0, s, A, B, bias, C, Nc);
    } else {
        if (act) hipLaunchKernelGGL((wmma_gemm_kernel<256, 1>), g, b, 0, s, A, B, bias, C, Nc);
        else     hipLaunchKernelGGL((wmma_gemm_kernel<256, 0>), g, b, 0, s, A, B, bias, C, Nc);
    }
}

extern "C" void kernel_launch(void* const* d_in, const int* in_sizes, int n_in,
                              void* d_out, int out_size, void* d_ws, size_t ws_size,
                              hipStream_t stream) {
    const int*   z      = (const int*)  d_in[0];
    const float* r_ij   = (const float*)d_in[1];
    const int*   idx_i  = (const int*)  d_in[2];
    const int*   idx_j  = (const int*)  d_in[3];
    const float* emb    = (const float*)d_in[4];
    const float* filt_W = (const float*)d_in[5];
    const float* filt_b = (const float*)d_in[6];
    const float* int_W1 = (const float*)d_in[7];
    const float* int_b1 = (const float*)d_in[8];
    const float* int_W2 = (const float*)d_in[9];
    const float* int_b2 = (const float*)d_in[10];
    const float* mix_W1 = (const float*)d_in[11];
    const float* mix_b1 = (const float*)d_in[12];
    const float* mix_W2 = (const float*)d_in[13];
    const float* mix_b2 = (const float*)d_in[14];
    const float* mux_W  = (const float*)d_in[15];

    const int N = in_sizes[0];
    const int E = in_sizes[2];
    const int Npad = ((N + 63) / 64) * 64;          // WMMA row-tile padding

    char* base = (char*)d_ws;
    size_t off = 0;
    auto alloc = [&](size_t bytes) -> char* {
        size_t o = (off + 255) & ~(size_t)255;
        off = o + bytes;
        return base + o;
    };

    float* q     = (float*)alloc((size_t)Npad * 128 * 4);
    float* mu    = (float*)alloc((size_t)Npad * 384 * 4);
    float* dq    = (float*)alloc((size_t)Npad * 128 * 4);
    float* dmu   = (float*)alloc((size_t)Npad * 384 * 4);
    float* hbuf  = (float*)alloc((size_t)Npad * 128 * 4);
    float* xbuf  = (float*)alloc((size_t)Npad * 384 * 4);
    float* mumix = (float*)alloc((size_t)Npad * 768 * 4);
    float* dlen  = (float*)alloc((size_t)E * 4);
    float* fcut  = (float*)alloc((size_t)E * 4);
    float* dir   = (float*)alloc((size_t)E * 12);
    unsigned short* wq    = (unsigned short*)alloc((size_t)Npad * 128 * 2);
    unsigned short* wh    = (unsigned short*)alloc((size_t)Npad * 128 * 2);
    unsigned short* wmu   = (unsigned short*)alloc((size_t)Npad * 384 * 2);
    unsigned short* wctx  = (unsigned short*)alloc((size_t)Npad * 256 * 2);
    unsigned short* wint1 = (unsigned short*)alloc((size_t)3 * 128 * 128 * 2);
    unsigned short* wint2 = (unsigned short*)alloc((size_t)3 * 128 * 384 * 2);
    unsigned short* wmix1 = (unsigned short*)alloc((size_t)3 * 256 * 128 * 2);
    unsigned short* wmix2 = (unsigned short*)alloc((size_t)3 * 128 * 384 * 2);
    unsigned short* wmux  = (unsigned short*)alloc((size_t)3 * 128 * 256 * 2);
    const size_t used = off;

    // zero the whole used workspace (pads, accumulators); ws is poisoned 0xAA.
    hipLaunchKernelGGL(zero_u32, dim3(2048), dim3(256), 0, stream,
                       (unsigned int*)d_ws, (used + 3) / 4);

    hipLaunchKernelGGL(geom_kernel, dim3((E + 255) / 256), dim3(256), 0, stream,
                       r_ij, dlen, fcut, dir, E);
    hipLaunchKernelGGL(embed_kernel, dim3((N * 128 + 255) / 256), dim3(256), 0, stream,
                       emb, z, q, N);

    // weights -> bf16 (once per call; tiny)
    hipLaunchKernelGGL(cvt_bf16, dim3(512), dim3(256), 0, stream, int_W1, wint1, (size_t)3 * 128 * 128);
    hipLaunchKernelGGL(cvt_bf16, dim3(512), dim3(256), 0, stream, int_W2, wint2, (size_t)3 * 128 * 384);
    hipLaunchKernelGGL(cvt_bf16, dim3(512), dim3(256), 0, stream, mix_W1, wmix1, (size_t)3 * 256 * 128);
    hipLaunchKernelGGL(cvt_bf16, dim3(512), dim3(256), 0, stream, mix_W2, wmix2, (size_t)3 * 128 * 384);
    hipLaunchKernelGGL(cvt_bf16, dim3(512), dim3(256), 0, stream, mux_W,  wmux,  (size_t)3 * 128 * 256);

    const dim3 nb((N * 128 + 255) / 256), tb(256);

    for (int l = 0; l < N_LAYERS; ++l) {
        // fresh scatter accumulators
        hipLaunchKernelGGL(zero_u32, dim3(1024), dim3(256), 0, stream,
                           (unsigned int*)dq, (size_t)Npad * 128);
        hipLaunchKernelGGL(zero_u32, dim3(1024), dim3(256), 0, stream,
                           (unsigned int*)dmu, (size_t)Npad * 384);

        // interaction: h = silu(q@W1+b1); x = h@W2+b2
        hipLaunchKernelGGL(cvt_bf16, dim3(1024), dim3(256), 0, stream, q, wq, (size_t)Npad * 128);
        launch_gemm(wq, wint1 + (size_t)l * 128 * 128, int_b1 + l * 128, hbuf,
                    Npad, 128, 128, 1, stream);
        hipLaunchKernelGGL(cvt_bf16, dim3(1024), dim3(256), 0, stream, hbuf, wh, (size_t)Npad * 128);
        launch_gemm(wh, wint2 + (size_t)l * 128 * 384, int_b2 + l * 384, xbuf,
                    Npad, 128, 384, 0, stream);

        // edge messages + scatter (one wave per edge)
        hipLaunchKernelGGL(edge_kernel, dim3(2048), dim3(128), 0, stream,
                           dlen, fcut, dir, idx_i, idx_j, xbuf, mu,
                           filt_W, filt_b, dq, dmu, E, l);

        hipLaunchKernelGGL(add_inplace, nb, tb, 0, stream, q, dq, N * 128);
        hipLaunchKernelGGL(add_inplace, dim3((N * 384 + 255) / 256), tb, 0, stream,
                           mu, dmu, N * 384);

        // mixing: mu_mix = mu @ muxW ; ctx = [q | ||mu_V||]
        hipLaunchKernelGGL(cvt_bf16, dim3(1024), dim3(256), 0, stream, mu, wmu, (size_t)Npad * 384);
        launch_gemm(wmu, wmux + (size_t)l * 128 * 256, nullptr, mumix,
                    3 * Npad, 128, 256, 0, stream);
        hipLaunchKernelGGL(ctx_kernel, nb, tb, 0, stream, q, mumix, wctx, N);
        launch_gemm(wctx, wmix1 + (size_t)l * 256 * 128, mix_b1 + l * 128, hbuf,
                    Npad, 256, 128, 1, stream);
        hipLaunchKernelGGL(cvt_bf16, dim3(1024), dim3(256), 0, stream, hbuf, wh, (size_t)Npad * 128);
        launch_gemm(wh, wmix2 + (size_t)l * 128 * 384, mix_b2 + l * 384, xbuf,
                    Npad, 128, 384, 0, stream);
        hipLaunchKernelGGL(update_kernel, nb, tb, 0, stream, q, mu, mumix, xbuf, N);
    }

    hipLaunchKernelGGL(copy_out_kernel, dim3((N * 512 + 255) / 256), dim3(256), 0, stream,
                       q, mu, (float*)d_out, N);
    (void)ws_size; (void)n_in; (void)out_size;
}
